// ChannelAttention_53266184405043
// MI455X (gfx1250) — matrix operand order repaired and checked by reading, not compiled, not run on hardware
//
#include <hip/hip_runtime.h>
#include <cfloat>

// ---------- types ----------
typedef __attribute__((ext_vector_type(16))) _Float16 v16h;
typedef __attribute__((ext_vector_type(8)))  float    v8f;
typedef __attribute__((ext_vector_type(4)))  float    v4f;

#define B_    32
#define C_    256
#define HW_   12544      // 112*112
#define HID_  32

#define SPLIT1 49        // pooling chunks per batch (49*256 = 12544 rows)
#define SPAN1  256       // spatial rows per pooling block
#define SPLIT2 98        // apply chunks per batch (98*128 = 12544 rows)
#define SPAN2  128

// =====================================================================
// Kernel 1: partial sum/max pooling.  grid = B_*SPLIT1, block = 256.
// Thread layout: g = tid%64 -> channel group (4 ch, b128), s = tid/64 ->
// spatial sub-lane.  Each 64-thread row reads a contiguous 1 KB segment.
// =====================================================================
__global__ __launch_bounds__(256) void pool_partial_kernel(
    const float* __restrict__ x, float* __restrict__ psum, float* __restrict__ pmax)
{
    const int b     = blockIdx.x / SPLIT1;
    const int chunk = blockIdx.x % SPLIT1;
    const int tid   = threadIdx.x;
    const int g     = tid & 63;   // channel group (4 channels)
    const int s     = tid >> 6;   // row sub-lane 0..3

    const v4f* xb = (const v4f*)(x + (size_t)b * HW_ * C_);
    v4f sum = {0.f, 0.f, 0.f, 0.f};
    v4f mx  = {-FLT_MAX, -FLT_MAX, -FLT_MAX, -FLT_MAX};

    const int base = chunk * SPAN1;
    #pragma unroll 8
    for (int i = 0; i < SPAN1 / 4; ++i) {
        const int r = base + s + 4 * i;
        v4f v = xb[(size_t)r * 64 + g];
        sum += v;
        mx.x = fmaxf(mx.x, v.x); mx.y = fmaxf(mx.y, v.y);
        mx.z = fmaxf(mx.z, v.z); mx.w = fmaxf(mx.w, v.w);
    }

    __shared__ v4f ssum[256];
    __shared__ v4f smax[256];
    ssum[tid] = sum;
    smax[tid] = mx;
    __syncthreads();

    if (tid < 64) {
        v4f s0 = ssum[tid], s1 = ssum[tid + 64], s2 = ssum[tid + 128], s3 = ssum[tid + 192];
        v4f m0 = smax[tid], m1 = smax[tid + 64], m2 = smax[tid + 128], m3 = smax[tid + 192];
        v4f S = (s0 + s1) + (s2 + s3);
        v4f M;
        M.x = fmaxf(fmaxf(m0.x, m1.x), fmaxf(m2.x, m3.x));
        M.y = fmaxf(fmaxf(m0.y, m1.y), fmaxf(m2.y, m3.y));
        M.z = fmaxf(fmaxf(m0.z, m1.z), fmaxf(m2.z, m3.z));
        M.w = fmaxf(fmaxf(m0.w, m1.w), fmaxf(m2.w, m3.w));
        const size_t o = ((size_t)b * SPLIT1 + chunk) * 64 + g;  // v4f units
        ((v4f*)psum)[o] = S;
        ((v4f*)pmax)[o] = M;
    }
}

// =====================================================================
// Kernel 2: fold the 49 partials (fixed order -> deterministic).
// grid = B_, block = 256 (one thread per channel).
// =====================================================================
__global__ __launch_bounds__(256) void pool_reduce_kernel(
    const float* __restrict__ psum, const float* __restrict__ pmax,
    float* __restrict__ pooled_avg, float* __restrict__ pooled_max)
{
    const int b = blockIdx.x;
    const int c = threadIdx.x;
    float s = 0.f, m = -FLT_MAX;
    for (int k = 0; k < SPLIT1; ++k) {
        const size_t o = ((size_t)b * SPLIT1 + k) * C_ + c;
        s += psum[o];
        m = fmaxf(m, pmax[o]);
    }
    pooled_avg[b * C_ + c] = s * (1.0f / (float)HW_);
    pooled_max[b * C_ + c] = m;
}

// =====================================================================
// Kernel 3: MLP + sigmoid with WMMA.  One block, 256 threads = 8 waves.
//   GEMM1: pool[32,256] @ w1[256,32] (+b1, relu) for both pools
//          -> 8 tiles of 16x16, one per wave, K = 8 steps of 32.
//   Hsum  = relu(avg) + relu(max)           (same W2 for both paths)
//   GEMM2: Hsum[32,32] @ w2[32,256] + 2*b2  -> sigmoid -> scale[32,256]
// Fragment layouts per CDNA5 ISA 7.12.2 (wave32).
// =====================================================================
__global__ __launch_bounds__(256) void mlp_wmma_kernel(
    const float* __restrict__ pooled_avg, const float* __restrict__ pooled_max,
    const float* __restrict__ w1, const float* __restrict__ b1,
    const float* __restrict__ w2, const float* __restrict__ b2,
    float* __restrict__ scale)
{
    __shared__ float    Hf32[2][HID_][HID_]; // relu outputs per pool (batch x hid)
    __shared__ _Float16 Hsum[HID_][HID_];

    const int tid  = threadIdx.x;
    const int wave = tid >> 5;
    const int lane = tid & 31;
    const int lm   = lane & 15;   // A row / B,C,D column within tile
    const int lh   = lane >> 4;   // half-select

    // ---------------- GEMM1 ----------------
    {
        const int pool = wave >> 2;          // 0 = avg, 1 = max
        const int mt   = (wave >> 1) & 1;    // batch tile
        const int nt   = wave & 1;           // hidden tile
        const float* __restrict__ P = pool ? pooled_max : pooled_avg;

        v8f acc = {0.f, 0.f, 0.f, 0.f, 0.f, 0.f, 0.f, 0.f};
        #pragma unroll
        for (int k0 = 0; k0 < C_; k0 += 32) {
            v16h a, bf;
            #pragma unroll
            for (int t = 0; t < 16; ++t) {
                // A (16x32 f16): lane lm = row, halves: K = 16*(t/8) + 8*lh + t%8
                const int ka = k0 + 16 * (t >> 3) + 8 * lh + (t & 7);
                a[t]  = (_Float16)P[(mt * 16 + lm) * C_ + ka];
                // B (32x16 f16): lane lm = col, halves: K = 16*(t/8) + 8*lh + t%8
                const int kb = k0 + 16 * (t >> 3) + 8 * lh + (t & 7);
                bf[t] = (_Float16)w1[kb * HID_ + (nt * 16 + lm)];
            }
            acc = __builtin_amdgcn_wmma_f32_16x16x32_f16(
                      false, a, false, bf, (short)0, acc, false, false);
        }
        const int   n    = nt * 16 + lm;
        const float bias = b1[n];
        #pragma unroll
        for (int r = 0; r < 8; ++r) {
            const int   m = mt * 16 + r + 8 * lh;   // C/D: M = r + 8*(lane/16)
            const float v = acc[r] + bias;
            Hf32[pool][m][n] = v > 0.f ? v : 0.f;   // relu
        }
    }
    __syncthreads();

    // Hsum = f16(Havg + Hmax): 1024 elements, 4 per thread
    #pragma unroll
    for (int i = 0; i < 4; ++i) {
        const int idx = tid + i * 256;
        const int m = idx >> 5, n = idx & 31;
        Hsum[m][n] = (_Float16)(Hf32[0][m][n] + Hf32[1][m][n]);
    }
    __syncthreads();

    // ---------------- GEMM2 ----------------
    #pragma unroll
    for (int i = 0; i < 4; ++i) {
        const int tile = wave * 4 + i;   // 0..31
        const int mt   = tile >> 4;      // batch tile
        const int nt   = tile & 15;      // channel tile

        v16h a, bf;
        #pragma unroll
        for (int t = 0; t < 16; ++t) {
            const int ka = 16 * (t >> 3) + 8 * lh + (t & 7);
            a[t]  = Hsum[mt * 16 + lm][ka];
            const int kb = 16 * (t >> 3) + 8 * lh + (t & 7);
            bf[t] = (_Float16)w2[kb * C_ + nt * 16 + lm];
        }
        v8f acc = {0.f, 0.f, 0.f, 0.f, 0.f, 0.f, 0.f, 0.f};
        acc = __builtin_amdgcn_wmma_f32_16x16x32_f16(
                  false, a, false, bf, (short)0, acc, false, false);

        const int   n     = nt * 16 + lm;
        const float bias2 = 2.0f * b2[n];   // b2 added once per MLP branch
        #pragma unroll
        for (int r = 0; r < 8; ++r) {
            const int   m = mt * 16 + r + 8 * lh;
            const float y = acc[r] + bias2;
            scale[m * C_ + n] = 1.0f / (1.0f + __expf(-y));
        }
    }
}

// =====================================================================
// Kernel 4: out = x * scale[b,c].  grid = B_*SPLIT2, block = 256.
// NT (last-use) loads of x, NT stores of out: neither is re-read.
// =====================================================================
__global__ __launch_bounds__(256) void apply_kernel(
    const float* __restrict__ x, const float* __restrict__ scale,
    float* __restrict__ out)
{
    const int b     = blockIdx.x / SPLIT2;
    const int chunk = blockIdx.x % SPLIT2;
    const int tid   = threadIdx.x;

    __shared__ v4f sscale[64];
    if (tid < 64) sscale[tid] = ((const v4f*)(scale + b * C_))[tid];
    __syncthreads();

    const int g = tid & 63;
    const int s = tid >> 6;
    const v4f sc = sscale[g];

    const v4f* xb = (const v4f*)(x   + (size_t)b * HW_ * C_);
    v4f*       ob = (v4f*)      (out + (size_t)b * HW_ * C_);

    const int base = chunk * SPAN2;
    #pragma unroll 8
    for (int i = 0; i < SPAN2 / 4; ++i) {
        const int   r   = base + s + 4 * i;
        const size_t o  = (size_t)r * 64 + g;
        v4f v = __builtin_nontemporal_load(&xb[o]);
        v *= sc;
        __builtin_nontemporal_store(v, &ob[o]);
    }
}

// =====================================================================
// Launch
// =====================================================================
extern "C" void kernel_launch(void* const* d_in, const int* in_sizes, int n_in,
                              void* d_out, int out_size, void* d_ws, size_t ws_size,
                              hipStream_t stream)
{
    const float* x  = (const float*)d_in[0];  // [32,112,112,256]
    const float* w1 = (const float*)d_in[1];  // [256,32]
    const float* b1 = (const float*)d_in[2];  // [32]
    const float* w2 = (const float*)d_in[3];  // [32,256]
    const float* b2 = (const float*)d_in[4];  // [256]
    float* out = (float*)d_out;

    // workspace layout (floats): ~3.3 MB total
    float* ws         = (float*)d_ws;
    float* psum       = ws;                                   // 32*49*256
    float* pmax       = psum + (size_t)B_ * SPLIT1 * C_;      // 32*49*256
    float* pooled_avg = pmax + (size_t)B_ * SPLIT1 * C_;      // 32*256
    float* pooled_max = pooled_avg + B_ * C_;                 // 32*256
    float* scale      = pooled_max + B_ * C_;                 // 32*256

    pool_partial_kernel<<<B_ * SPLIT1, 256, 0, stream>>>(x, psum, pmax);
    pool_reduce_kernel<<<B_, 256, 0, stream>>>(psum, pmax, pooled_avg, pooled_max);
    mlp_wmma_kernel<<<1, 256, 0, stream>>>(pooled_avg, pooled_max, w1, b1, w2, b2, scale);
    apply_kernel<<<B_ * SPLIT2, 256, 0, stream>>>(x, scale, out);
}
